// ARCStacked_12558484374037
// MI455X (gfx1250) — matrix operand order repaired and checked by reading, not compiled, not run on hardware
//
#include <hip/hip_runtime.h>
#include <hip/hip_bf16.h>
#include <math.h>

// ---------------- problem constants ----------------
constexpr int BATCH = 2;
constexpr int SEQ   = 1024;
constexpr int ST    = 64;
constexpr int T     = SEQ + 2 * ST;   // 1152
constexpr int H     = 12;
constexpr int DI    = 768;
constexpr int DK    = 64;
constexpr int DV    = 64;
constexpr int BH    = BATCH * H;      // 24
constexpr int NT_T  = T / 16;         // 72 column tiles of 16
constexpr float SCALE_F = 0.125f;     // 1/sqrt(64)

typedef __attribute__((ext_vector_type(16))) _Float16 v16h;
typedef __attribute__((ext_vector_type(8)))  _Float16 v8h;
typedef __attribute__((ext_vector_type(4)))  _Float16 v4h;
typedef __attribute__((ext_vector_type(8)))  float    v8f;

__device__ __forceinline__ v8f wmma16(v16h a, v16h b, v8f c) {
    // D = A(16x32 f16) * B(32x16 f16) + C(16x16 f32)
    return __builtin_amdgcn_wmma_f32_16x16x32_f16(
        false, a, false, b, (short)0, c, false, false);
}

// Load a B fragment (32x16, K along 32): for this lane the 16 needed f16
// elements are contiguous at `p` (K(e) = hi*16 + e).
__device__ __forceinline__ v16h load_bfrag(const _Float16* p) {
    const v8h* q = (const v8h*)p;
    v8h lo = q[0], hf = q[1];
    v16h r;
#pragma unroll
    for (int j = 0; j < 8; ++j) { r[j] = lo[j]; r[8 + j] = hf[j]; }
    return r;
}

// ---------------- kernel 0a: cope_emb [H][DK][T] f32 -> cope_t [H][T][DK] f16
__global__ __launch_bounds__(256) void cope_cvt_kernel(
    const float* __restrict__ cope, _Float16* __restrict__ ct) {
    int idx = blockIdx.x * 256 + threadIdx.x;
    if (idx >= H * T * DK) return;
    int dk = idx % DK;
    int t  = (idx / DK) % T;
    int h  = idx / (DK * T);
    ct[idx] = (_Float16)cope[((size_t)h * DK + dk) * T + t];
}

// ---------------- kernel 0b: weights [H][DI][64] f32 -> [H][64][DI] f16 -----
// One matrix per blockIdx.y (q,k,v,qs,ks,vs); d is fastest per thread so the
// f16 stores coalesce; reused 72x per (b,h) afterwards so one-time cost.
__global__ __launch_bounds__(256) void w_cvt_kernel(
    const float* __restrict__ wq,  const float* __restrict__ wk,
    const float* __restrict__ wv,  const float* __restrict__ wqs,
    const float* __restrict__ wks, const float* __restrict__ wvs,
    _Float16* __restrict__ wT) {
    const float* src;
    switch (blockIdx.y) {
        case 0: src = wq;  break;
        case 1: src = wk;  break;
        case 2: src = wv;  break;
        case 3: src = wqs; break;
        case 4: src = wks; break;
        default: src = wvs; break;
    }
    const size_t MSZ = (size_t)H * 64 * DI;
    size_t idx = (size_t)blockIdx.x * 256 + threadIdx.x;
    if (idx >= MSZ) return;
    int d = idx % DI;
    int n = (idx / DI) % 64;
    int h = idx / (DI * 64);
    wT[blockIdx.y * MSZ + idx] = (_Float16)src[((size_t)h * DI + d) * 64 + n];
}

// ---------------- kernel 1: QKV projection ----------------
// grid = BH * NT_T blocks, 128 threads (4 waves). Each block: one 16-row t tile
// of one (b,h): Q,K (row-major f16 [bh][t][64]) and V transposed ([bh][dv][T]).
__global__ __launch_bounds__(128) void qkv_kernel(
    const float* __restrict__ x, const _Float16* __restrict__ wT,
    _Float16* __restrict__ Qh, _Float16* __restrict__ Kh,
    _Float16* __restrict__ Vt) {
    int blk   = blockIdx.x;
    int ttile = blk % NT_T;
    int bh    = blk / NT_T;
    int b     = bh / H;
    int h     = bh % H;
    int t0    = ttile * 16;
    bool is_state = (ttile < ST / 16) || (ttile >= (T - ST) / 16);

    __shared__ _Float16 xs[16][776];   // padded rows: distinct banks on A gather

    // stage X tile (16 x 768) into LDS as f16, vectorized float4 -> v4h
    const float* xbase = x + (size_t)b * T * DI;
    for (int idx = threadIdx.x; idx < 16 * (DI / 4); idx += 128) {
        int m  = idx / (DI / 4);
        int d4 = idx % (DI / 4);
        float4 v = ((const float4*)(xbase + (size_t)(t0 + m) * DI))[d4];
        v4h o;
        o[0] = (_Float16)v.x; o[1] = (_Float16)v.y;
        o[2] = (_Float16)v.z; o[3] = (_Float16)v.w;
        *(v4h*)&xs[m][d4 * 4] = o;
    }
    __syncthreads();

    int lane = threadIdx.x & 31;
    int m    = lane & 15;
    int hi   = lane >> 4;
    int nt   = threadIdx.x >> 5;   // output column tile 0..3 (16 cols each)
    int col  = nt * 16 + m;

    const size_t MSZ = (size_t)H * 64 * DI;
    const int sel = is_state ? 3 : 0;
    const _Float16* Wq = wT + (sel + 0) * MSZ + ((size_t)h * 64 + col) * DI;
    const _Float16* Wk = wT + (sel + 1) * MSZ + ((size_t)h * 64 + col) * DI;
    const _Float16* Wv = wT + (sel + 2) * MSZ + ((size_t)h * 64 + col) * DI;

    v8f cq = {}, ck = {}, cv = {};
    for (int ks = 0; ks < DI / 32; ++ks) {
        // A fragment from LDS: K(e) = hi*16 + e
        v16h a;
        {
            const v8h* p0 = (const v8h*)&xs[m][ks * 32 + hi * 16];
            const v8h* p1 = (const v8h*)&xs[m][ks * 32 + hi * 16 + 8];
            v8h lo = *p0, hf = *p1;
#pragma unroll
            for (int j = 0; j < 8; ++j) { a[j] = lo[j]; a[8 + j] = hf[j]; }
        }
        int dbase = ks * 32 + hi * 16;          // contiguous f16 B fragments
        v16h bq = load_bfrag(Wq + dbase);
        v16h bk = load_bfrag(Wk + dbase);
        v16h bv = load_bfrag(Wv + dbase);
        cq = wmma16(a, bq, cq);
        ck = wmma16(a, bk, ck);
        cv = wmma16(a, bv, cv);
    }

    size_t base = (size_t)bh * T * 64;
#pragma unroll
    for (int i = 0; i < 8; ++i) {
        int row = t0 + i + hi * 8;
        Qh[base + (size_t)row * 64 + col] = (_Float16)cq[i];
        Kh[base + (size_t)row * 64 + col] = (_Float16)ck[i];
        Vt[base + (size_t)col * T + row] = (_Float16)cv[i];   // transposed store
    }
}

// ---------------- kernel 2: fused CoPE attention ----------------
// grid = BH * NT_T blocks (one 16-query tile each), 256 threads (8 waves).
__global__ __launch_bounds__(256) void attn_kernel(
    const _Float16* __restrict__ Qh, const _Float16* __restrict__ Kh,
    const _Float16* __restrict__ Vt, const _Float16* __restrict__ ct,
    float* __restrict__ out) {
    extern __shared__ float smem[];
    constexpr int LDW = T + 4;         // 1156 f32: pad for column-gather banks
    constexpr int PW  = T + 8;         // 1160 f16: pad for Phase C A gather
    float* Slog = smem;                // [16][LDW] logits -> scores -> exp
    float* Sli  = smem + 16 * LDW;     // [16][LDW] cope logits (dead after B1)
    _Float16* Ph = (_Float16*)Sli;     // [16][PW] f16 probs (reuses Sli space)

    int blk  = blockIdx.x;
    int qt   = blk % NT_T;
    int bh   = blk / NT_T;
    int h    = bh % H;
    int t0   = qt * 16;
    int wave = threadIdx.x >> 5;
    int lane = threadIdx.x & 31;
    int m    = lane & 15;
    int hi   = lane >> 4;

    // Q A-fragments for the two 32-wide K-steps of DK=64
    v16h aq[2];
    {
        const _Float16* qb = Qh + ((size_t)bh * T + t0 + m) * 64;
#pragma unroll
        for (int ks = 0; ks < 2; ++ks) {
            const v8h* p0 = (const v8h*)(qb + ks * 32 + hi * 16);
            const v8h* p1 = (const v8h*)(qb + ks * 32 + hi * 16 + 8);
            v8h lo = *p0, hf = *p1;
#pragma unroll
            for (int j = 0; j < 8; ++j) { aq[ks][j] = lo[j]; aq[ks][8 + j] = hf[j]; }
        }
    }

    // ---- Phase A: logits tile (Q.K^T) and cope-logits tile (Q.cope) into LDS
    for (int nt = wave; nt < NT_T; nt += 8) {
        int tk = nt * 16 + m;                  // key index / cope position
        v8f c1 = {}, c2 = {};
#pragma unroll
        for (int ks = 0; ks < 2; ++ks) {
            v16h bk = load_bfrag(Kh + ((size_t)bh * T + tk) * 64 + ks * 32 + hi * 16);
            v16h bc = load_bfrag(ct + ((size_t)h  * T + tk) * 64 + ks * 32 + hi * 16);
            c1 = wmma16(aq[ks], bk, c1);
            c2 = wmma16(aq[ks], bc, c2);
        }
#pragma unroll
        for (int i = 0; i < 8; ++i) {
            Slog[(i + hi * 8) * LDW + nt * 16 + m] = c1[i];
            Sli [(i + hi * 8) * LDW + nt * 16 + m] = c2[i];
        }
    }
    __syncthreads();

    // ---- Phase B1: sigmoid -> suffix cumsum -> bias -> masked scores + rowmax
    float rmax[2];
#pragma unroll
    for (int rr = 0; rr < 2; ++rr) {
        int r  = wave + rr * 8;                // rows 0..15 over 8 waves
        int qg = t0 + r;
        float* srow = Slog + (size_t)r * LDW;
        const float* lrow = Sli + (size_t)r * LDW;

        float carry = 0.f;
        float rowmax = -INFINITY;
        for (int c = T / 32 - 1; c >= 0; --c) {  // descending key chunks of 32
            int k = c * 32 + lane;
            float lg = srow[k];
            float g = 1.f / (1.f + __expf(-lg));
            // wave-level suffix scan: s[l] = sum_{j>=l} g[j]
            float s = g;
#pragma unroll
            for (int off = 1; off < 32; off <<= 1) {
                float tsh = __shfl_down(s, off, 32);
                if (lane + off < 32) s += tsh;
            }
            float pos = fminf(s + carry, (float)(T - 1));
            carry += __shfl(s, 0, 32);         // chunk total
            float pf  = floorf(pos);
            float wgt = pos - pf;
            int ipf = (int)pf;
            int ipc = (int)ceilf(pos);
            if (ipc > T - 1) ipc = T - 1;
            float bias = lrow[ipc] * wgt + lrow[ipf] * (1.f - wgt);
            float sc = lg * SCALE_F + bias;
            if (k > qg) sc = -INFINITY;        // causal mask
            srow[k] = sc;
            rowmax = fmaxf(rowmax, sc);
        }
#pragma unroll
        for (int off = 16; off > 0; off >>= 1)
            rowmax = fmaxf(rowmax, __shfl_xor(rowmax, off, 32));
        rmax[rr] = rowmax;
    }
    __syncthreads();   // Sli fully consumed; its space becomes Ph (f16 probs)

    // ---- Phase B2: softmax normalize, emit f16 probs laid out for A-frags
#pragma unroll
    for (int rr = 0; rr < 2; ++rr) {
        int r = wave + rr * 8;
        float* srow = Slog + (size_t)r * LDW;
        float sum = 0.f;
        for (int c = 0; c < T / 32; ++c) {
            int k = c * 32 + lane;
            float e = __expf(srow[k] - rmax[rr]);
            srow[k] = e;
            sum += e;
        }
#pragma unroll
        for (int off = 16; off > 0; off >>= 1)
            sum += __shfl_xor(sum, off, 32);
        float inv = 1.f / sum;
        for (int c = 0; c < T / 32; ++c) {
            int k = c * 32 + lane;
            Ph[(size_t)r * PW + k] = (_Float16)(srow[k] * inv);
        }
    }
    __syncthreads();

    // ---- Phase C: att = P(16x1152) . V(1152x64), waves 0..3 own one dv tile
    if (wave < 4) {
        int dvc = wave * 16 + m;
        const _Float16* vb = Vt + ((size_t)bh * 64 + dvc) * T;   // V transposed
        v8f c = {};
        for (int ks = 0; ks < T / 32; ++ks) {
            // A fragment: two contiguous 16B LDS loads from f16 probs
            const v8h* pa0 = (const v8h*)&Ph[(size_t)m * PW + ks * 32 + hi * 16];
            const v8h* pa1 = (const v8h*)&Ph[(size_t)m * PW + ks * 32 + hi * 16 + 8];
            v8h lo = *pa0, hf = *pa1;
            v16h a;
#pragma unroll
            for (int j = 0; j < 8; ++j) { a[j] = lo[j]; a[8 + j] = hf[j]; }
            v16h bv = load_bfrag(vb + ks * 32 + hi * 16);
            c = wmma16(a, bv, c);
        }
#pragma unroll
        for (int i = 0; i < 8; ++i)
            out[((size_t)bh * T + t0 + i + hi * 8) * 64 + dvc] = c[i];
    }
}

// ---------------- launcher ----------------
extern "C" void kernel_launch(void* const* d_in, const int* in_sizes, int n_in,
                              void* d_out, int out_size, void* d_ws, size_t ws_size,
                              hipStream_t stream) {
    const float* x    = (const float*)d_in[0];
    const float* wq   = (const float*)d_in[1];
    const float* wk   = (const float*)d_in[2];
    const float* wv   = (const float*)d_in[3];
    const float* wqs  = (const float*)d_in[4];
    const float* wks  = (const float*)d_in[5];
    const float* wvs  = (const float*)d_in[6];
    const float* cope = (const float*)d_in[7];

    _Float16* ws = (_Float16*)d_ws;
    const size_t QSZ = (size_t)BH * T * 64;        // halves per Q/K/V buffer
    const size_t CSZ = (size_t)H * T * DK;         // cope_t halves
    const size_t MSZ = (size_t)H * 64 * DI;        // halves per weight matrix
    _Float16* Qh = ws;
    _Float16* Kh = ws + QSZ;
    _Float16* Vt = ws + 2 * QSZ;
    _Float16* ct = ws + 3 * QSZ;                   // [H][T][DK] f16
    _Float16* wT = ws + 3 * QSZ + CSZ;             // [6][H][64][DI] f16

    cope_cvt_kernel<<<(H * T * DK + 255) / 256, 256, 0, stream>>>(cope, ct);
    {
        dim3 grid((unsigned)((MSZ + 255) / 256), 6);
        w_cvt_kernel<<<grid, 256, 0, stream>>>(wq, wk, wv, wqs, wks, wvs, wT);
    }
    qkv_kernel<<<BH * NT_T, 128, 0, stream>>>(x, wT, Qh, Kh, Vt);

    const size_t smem = (size_t)2 * 16 * (T + 4) * sizeof(float);  // 147,968 B
    (void)hipFuncSetAttribute((const void*)attn_kernel,
                              hipFuncAttributeMaxDynamicSharedMemorySize,
                              (int)smem);
    attn_kernel<<<BH * NT_T, 256, smem, stream>>>(Qh, Kh, Vt, ct, (float*)d_out);
}
